// VQVAE_38697655337581
// MI455X (gfx1250) — matrix operand order repaired and checked by reading, not compiled, not run on hardware
//
#include <hip/hip_runtime.h>
#include <hip/hip_bf16.h>
#include <math.h>

typedef __attribute__((ext_vector_type(16))) _Float16 v16h;
typedef __attribute__((ext_vector_type(8)))  float    v8f;
typedef __attribute__((ext_vector_type(4)))  unsigned u32x4;
typedef _Float16 h16;

#define NT 4   // N-tiles per wave
#define MT 2   // M-tiles per wave  -> 8 accumulators, B shared across M

// =====================================================================
// enc1: direct conv 3->64, k4 s2 p1, +ReLU.  fp32 NCHW in -> f16 NHWC out
// =====================================================================
__global__ __launch_bounds__(256)
void k_enc1(const float* __restrict__ x, const float* __restrict__ w,
            const float* __restrict__ b, h16* __restrict__ out)
{
  long tid = (long)blockIdx.x * blockDim.x + threadIdx.x;
  if (tid >= (long)64 * 64 * 64 * 64) return;
  int co = (int)(tid & 63);
  long m  = tid >> 6;
  int ow = (int)(m & 63);
  int oh = (int)((m >> 6) & 63);
  int bb = (int)(m >> 12);
  int ih0 = oh * 2 - 1, iw0 = ow * 2 - 1;
  float acc = b[co];
  #pragma unroll
  for (int ci = 0; ci < 3; ++ci) {
    #pragma unroll
    for (int kh = 0; kh < 4; ++kh) {
      int ih = ih0 + kh;
      if ((unsigned)ih >= 128u) continue;
      #pragma unroll
      for (int kw = 0; kw < 4; ++kw) {
        int iw = iw0 + kw;
        if ((unsigned)iw >= 128u) continue;
        acc += x[(((long)bb * 3 + ci) * 128 + ih) * 128 + iw] *
               w[(((long)co * 3 + ci) * 4 + kh) * 4 + kw];
      }
    }
  }
  acc = acc > 0.f ? acc : 0.f;
  out[tid] = (h16)acc;
}

// =====================================================================
// Weight transforms -> WMMA B-fragment-swizzled layout:
//   frag[(ktile*Ntiles + ntile)*32*16 + lane*16 + e]
//     = Wkn[k][n],  k = ktile*32 + 16*(e>>3) + 8*(lane>>4) + (e&7),  n = ntile*16 + (lane&15)
// =====================================================================
__global__ __launch_bounds__(256)
void k_wt_conv_frag(const float* __restrict__ w, h16* __restrict__ o,
                    int Cout, int Cin, int KH, int KW)
{
  long total = (long)KH * KW * Cin * Cout;
  long tid = (long)blockIdx.x * blockDim.x + threadIdx.x;
  if (tid >= total) return;
  int Ntiles = Cout >> 4;
  int e    = (int)(tid & 15);
  int lane = (int)((tid >> 4) & 31);
  long rest = tid >> 9;
  int ntile = (int)(rest % Ntiles);
  int ktile = (int)(rest / Ntiles);
  int k = ktile * 32 + ((e >> 3) << 4) + ((lane >> 4) << 3) + (e & 7);
  int n = ntile * 16 + (lane & 15);
  int ci = k % Cin;
  int t  = k / Cin;
  int kh = t / KW, kw = t - kh * KW;
  o[tid] = (h16)w[(((long)n * Cin + ci) * KH + kh) * KW + kw];
}

__global__ __launch_bounds__(256)
void k_wt_deconv_frag(const float* __restrict__ w, h16* __restrict__ o,
                      int Cin, int Cout, int KH, int KW,
                      int kh0, int kh1, int kw0, int kw1)
{
  long total = (long)4 * Cin * Cout;
  long tid = (long)blockIdx.x * blockDim.x + threadIdx.x;
  if (tid >= total) return;
  int Ntiles = Cout >> 4;
  int e    = (int)(tid & 15);
  int lane = (int)((tid >> 4) & 31);
  long rest = tid >> 9;
  int ntile = (int)(rest % Ntiles);
  int ktile = (int)(rest / Ntiles);
  int k = ktile * 32 + ((e >> 3) << 4) + ((lane >> 4) << 3) + (e & 7);
  int n = ntile * 16 + (lane & 15);
  int ci = k % Cin;
  int t  = k / Cin;
  int kh = (t >> 1) ? kh1 : kh0;
  int kw = (t & 1)  ? kw1 : kw0;
  o[tid] = (h16)w[(((long)ci * Cout + n) * KH + kh) * KW + kw];
}

__global__ __launch_bounds__(256)
void k_embT2_frag(const float* __restrict__ emb, h16* __restrict__ o)
{
  int tid = blockIdx.x * blockDim.x + threadIdx.x;
  if (tid >= 128 * 512) return;
  const int Ntiles = 32;
  int e    = tid & 15;
  int lane = (tid >> 4) & 31;
  int rest = tid >> 9;
  int ntile = rest % Ntiles;
  int ktile = rest / Ntiles;
  int k = ktile * 32 + ((e >> 3) << 4) + ((lane >> 4) << 3) + (e & 7);
  int n = ntile * 16 + (lane & 15);
  o[tid] = (h16)(-2.0f * emb[(long)n * 128 + k]);
}

__global__ __launch_bounds__(256)
void k_enorm(const float* __restrict__ emb, float* __restrict__ o)
{
  int n = blockIdx.x * blockDim.x + threadIdx.x;
  if (n >= 512) return;
  float s = 0.f;
  #pragma unroll 8
  for (int c = 0; c < 128; ++c) { float v = emb[(long)n * 128 + c]; s += v * v; }
  o[n] = s;
}

// ---------------------------------------------------------------------
union AFrag { v16h v; u32x4 q[2]; };
union BFrag { v16h v; u32x4 q[2]; };

static __device__ __forceinline__ void load_bfrag(BFrag& b, const h16* wfrag,
                                                  long ktile, int Ntiles,
                                                  int ntile, int lane)
{
  const u32x4* p = reinterpret_cast<const u32x4*>(
      wfrag + (((ktile * Ntiles) + ntile) * 32 + lane) * 16);
  b.q[0] = p[0];
  b.q[1] = p[1];
}

// =====================================================================
// WMMA implicit-GEMM conv.  Wave = MT(2) M-tiles x NT(4) N-tiles = 8 acc.
// Per K-step: 2*MT A loads + 2*NT B loads + MT*NT wmma (B shared over M).
// block = 128 (4 waves); wave covers 32 rows; grid = (M/128, Ntiles/NT).
// =====================================================================
__global__ __launch_bounds__(128)
void k_wmma_conv(const h16* __restrict__ in, const h16* __restrict__ wfrag,
                 const float* __restrict__ bias, h16* __restrict__ out,
                 int Hin, int Win, int cinShift,
                 int Hout, int Wout, int Cout,
                 int KH, int KW, int stride, int pad,
                 int Ntiles, int relu)
{
  int wave = threadIdx.x >> 5;
  int lane = threadIdx.x & 31;
  int nt0  = blockIdx.y * NT;
  int m0   = blockIdx.x * (MT * 64) + wave * (MT * 16);
  int r  = lane & 15;
  int hg = lane >> 4;
  int Cin = 1 << cinShift;

  // decode the MT im2col rows this lane loads (hoisted)
  int bAv[MT], ihB[MT], iwB[MT];
  #pragma unroll
  for (int i = 0; i < MT; ++i) {
    int mA = m0 + i * 16 + r;
    int owA = mA % Wout;
    int tA  = mA / Wout;
    int ohA = tA % Hout;
    bAv[i]  = tA / Hout;
    ihB[i]  = ohA * stride - pad;
    iwB[i]  = owA * stride - pad;
  }
  const u32x4 zero4 = {0u, 0u, 0u, 0u};

  v8f acc[MT][NT];
  #pragma unroll
  for (int i = 0; i < MT; ++i)
    #pragma unroll
    for (int j = 0; j < NT; ++j) acc[i][j] = (v8f){};

  long ktile = 0;
  for (int kh = 0; kh < KH; ++kh) {
    for (int kw = 0; kw < KW; ++kw) {
      const h16* rowp[MT];
      bool ok[MT];
      #pragma unroll
      for (int i = 0; i < MT; ++i) {
        int ih = ihB[i] + kh;
        int iw = iwB[i] + kw;
        bool okh = (unsigned)ih < (unsigned)Hin;
        bool okw = (unsigned)iw < (unsigned)Win;
        ok[i] = okh && okw;
        int ihc = okh ? ih : 0;
        int iwc = okw ? iw : 0;
        rowp[i] = in + ((((long)bAv[i] * Hin + ihc) * Win + iwc) << cinShift)
                  + (hg << 3);
      }
      for (int c0 = 0; c0 < Cin; c0 += 32, ++ktile) {
        AFrag a[MT];
        #pragma unroll
        for (int i = 0; i < MT; ++i) {
          u32x4 t0 = *reinterpret_cast<const u32x4*>(rowp[i] + c0);
          u32x4 t1 = *reinterpret_cast<const u32x4*>(rowp[i] + c0 + 16);
          a[i].q[0] = ok[i] ? t0 : zero4;
          a[i].q[1] = ok[i] ? t1 : zero4;
        }
        #pragma unroll
        for (int j = 0; j < NT; ++j) {
          BFrag b;
          load_bfrag(b, wfrag, ktile, Ntiles, nt0 + j, lane);
          #pragma unroll
          for (int i = 0; i < MT; ++i)
            acc[i][j] = __builtin_amdgcn_wmma_f32_16x16x32_f16(
                false, a[i].v, false, b.v, (short)0, acc[i][j], false, false);
        }
      }
    }
  }
  #pragma unroll
  for (int i = 0; i < MT; ++i) {
    long obase = ((long)(m0 + i * 16 + (hg << 3))) * Cout;
    #pragma unroll
    for (int j = 0; j < NT; ++j) {
      int nB = ((nt0 + j) << 4) + r;
      float bz = bias[nB];
      #pragma unroll
      for (int v = 0; v < 8; ++v) {
        float xv = acc[i][j][v] + bz;
        if (relu) xv = xv > 0.f ? xv : 0.f;
        out[obase + (long)v * Cout + nB] = (h16)xv;
      }
    }
  }
}

// =====================================================================
// WMMA phase deconv (k4 s2 p1): 2x2 taps, K = 4*Cin; same tiling.
// =====================================================================
__global__ __launch_bounds__(128)
void k_wmma_deconv(const h16* __restrict__ in, const h16* __restrict__ wfrag,
                   const float* __restrict__ bias, h16* __restrict__ out,
                   int Hin, int Win, int cinShift,
                   int Hout, int Wout, int Cout,
                   int pa, int pb, int kh0, int kh1, int kw0, int kw1,
                   int Ntiles, int relu)
{
  int wave = threadIdx.x >> 5;
  int lane = threadIdx.x & 31;
  int nt0  = blockIdx.y * NT;
  int m0   = blockIdx.x * (MT * 64) + wave * (MT * 16);
  int r  = lane & 15;
  int hg = lane >> 4;
  int Cin = 1 << cinShift;
  int H2 = Hout >> 1, W2 = Wout >> 1;

  int  bAv[MT];
  int  ihv[MT][2], iwv[MT][2];
  bool vh[MT][2],  vw[MT][2];
  #pragma unroll
  for (int i = 0; i < MT; ++i) {
    int mA = m0 + i * 16 + r;
    int oww = mA % W2;
    int tA  = mA / W2;
    int ohh = tA % H2;
    bAv[i]  = tA / H2;
    int oh = 2 * ohh + pa, ow = 2 * oww + pb;
    int nh[2] = { oh + 1 - kh0, oh + 1 - kh1 };
    int nw[2] = { ow + 1 - kw0, ow + 1 - kw1 };
    #pragma unroll
    for (int q = 0; q < 2; ++q) {
      int A = nh[q] >> 1;  vh[i][q] = (nh[q] >= 0) && (A < Hin);  ihv[i][q] = vh[i][q] ? A : 0;
      int C = nw[q] >> 1;  vw[i][q] = (nw[q] >= 0) && (C < Win);  iwv[i][q] = vw[i][q] ? C : 0;
    }
  }
  const u32x4 zero4 = {0u, 0u, 0u, 0u};

  v8f acc[MT][NT];
  #pragma unroll
  for (int i = 0; i < MT; ++i)
    #pragma unroll
    for (int j = 0; j < NT; ++j) acc[i][j] = (v8f){};

  long ktile = 0;
  #pragma unroll
  for (int th = 0; th < 2; ++th) {
    #pragma unroll
    for (int tw = 0; tw < 2; ++tw) {
      const h16* rowp[MT];
      bool ok[MT];
      #pragma unroll
      for (int i = 0; i < MT; ++i) {
        ok[i] = vh[i][th] && vw[i][tw];
        rowp[i] = in
            + ((((long)bAv[i] * Hin + ihv[i][th]) * Win + iwv[i][tw]) << cinShift)
            + (hg << 3);
      }
      for (int c0 = 0; c0 < Cin; c0 += 32, ++ktile) {
        AFrag a[MT];
        #pragma unroll
        for (int i = 0; i < MT; ++i) {
          u32x4 t0 = *reinterpret_cast<const u32x4*>(rowp[i] + c0);
          u32x4 t1 = *reinterpret_cast<const u32x4*>(rowp[i] + c0 + 16);
          a[i].q[0] = ok[i] ? t0 : zero4;
          a[i].q[1] = ok[i] ? t1 : zero4;
        }
        #pragma unroll
        for (int j = 0; j < NT; ++j) {
          BFrag b;
          load_bfrag(b, wfrag, ktile, Ntiles, nt0 + j, lane);
          #pragma unroll
          for (int i = 0; i < MT; ++i)
            acc[i][j] = __builtin_amdgcn_wmma_f32_16x16x32_f16(
                false, a[i].v, false, b.v, (short)0, acc[i][j], false, false);
        }
      }
    }
  }
  // epilogue: 8 rows per lane stay inside one output row (W2 >= 32)
  #pragma unroll
  for (int i = 0; i < MT; ++i) {
    int mm0 = m0 + i * 16 + (hg << 3);
    int oww2 = mm0 % W2;
    int t2   = mm0 / W2;
    int ohh2 = t2 % H2;
    int b2   = t2 / H2;
    int oh2 = 2 * ohh2 + pa;
    long obase = (((long)b2 * Hout + oh2) * Wout + (2 * oww2 + pb)) * Cout;
    #pragma unroll
    for (int j = 0; j < NT; ++j) {
      int nB = ((nt0 + j) << 4) + r;
      float bz = bias[nB];
      #pragma unroll
      for (int v = 0; v < 8; ++v) {
        float xv = acc[i][j][v] + bz;
        if (relu) xv = xv > 0.f ? xv : 0.f;
        out[obase + (long)(2 * v) * Cout + nB] = (h16)xv;
      }
    }
  }
}

// =====================================================================
// VQ scores: S[m][n] = -2 * z_e[m].emb[n] + |emb[n]|^2  (fp32)
// GEMM M=65536, K=128, N=512; wave = 2 M-tiles x 4 N-tiles.
// =====================================================================
__global__ __launch_bounds__(128)
void k_vq_scores(const h16* __restrict__ ze, const h16* __restrict__ embF,
                 const float* __restrict__ enorm, float* __restrict__ scores)
{
  int wave = threadIdx.x >> 5;
  int lane = threadIdx.x & 31;
  int nt0  = blockIdx.y * NT;          // grid.y = 8 -> ntiles 0..31
  int m0   = blockIdx.x * (MT * 64) + wave * (MT * 16);
  int r  = lane & 15;
  int hg = lane >> 4;
  const h16* arow[MT];
  #pragma unroll
  for (int i = 0; i < MT; ++i)
    arow[i] = ze + (long)(m0 + i * 16 + r) * 128 + (hg << 3);

  v8f acc[MT][NT];
  #pragma unroll
  for (int i = 0; i < MT; ++i)
    #pragma unroll
    for (int j = 0; j < NT; ++j) acc[i][j] = (v8f){};

  #pragma unroll
  for (int kt = 0; kt < 4; ++kt) {
    AFrag a[MT];
    #pragma unroll
    for (int i = 0; i < MT; ++i) {
      a[i].q[0] = *reinterpret_cast<const u32x4*>(arow[i] + kt * 32);
      a[i].q[1] = *reinterpret_cast<const u32x4*>(arow[i] + kt * 32 + 16);
    }
    #pragma unroll
    for (int j = 0; j < NT; ++j) {
      BFrag b;
      load_bfrag(b, embF, kt, 32, nt0 + j, lane);
      #pragma unroll
      for (int i = 0; i < MT; ++i)
        acc[i][j] = __builtin_amdgcn_wmma_f32_16x16x32_f16(
            false, a[i].v, false, b.v, (short)0, acc[i][j], false, false);
    }
  }
  #pragma unroll
  for (int i = 0; i < MT; ++i) {
    long obase = (long)(m0 + i * 16 + (hg << 3)) * 512;
    #pragma unroll
    for (int j = 0; j < NT; ++j) {
      int nB = ((nt0 + j) << 4) + r;
      float bz = enorm[nB];
      #pragma unroll
      for (int v = 0; v < 8; ++v)
        scores[obase + (long)v * 512 + nB] = acc[i][j][v] + bz;
    }
  }
}

// =====================================================================
// per-row argmin over 512 codes + gather z_q = emb[idx] (f16); wave32/row
// =====================================================================
__global__ __launch_bounds__(256)
void k_argmin_gather(const float* __restrict__ scores, const float* __restrict__ emb,
                     h16* __restrict__ zq, int M)
{
  int wave = threadIdx.x >> 5;
  int lane = threadIdx.x & 31;
  int m = blockIdx.x * 8 + wave;
  if (m >= M) return;
  const float* row = scores + (long)m * 512;
  float bv = 3.4e38f; int bi = 0;
  #pragma unroll
  for (int i = 0; i < 16; ++i) {
    int n = lane + i * 32;
    float v = row[n];
    if (v < bv) { bv = v; bi = n; }
  }
  #pragma unroll
  for (int off = 16; off > 0; off >>= 1) {
    float ov = __shfl_xor(bv, off, 32);
    int   oi = __shfl_xor(bi, off, 32);
    if (ov < bv || (ov == bv && oi < bi)) { bv = ov; bi = oi; }
  }
  const float* e = emb + (long)bi * 128;
  #pragma unroll
  for (int i = 0; i < 4; ++i) {
    int c = lane + i * 32;
    zq[(long)m * 128 + c] = (h16)e[c];
  }
}

// =====================================================================
// dec3: direct conv 64->3 k3 s1 p1 + bias + tanh.  f16 NHWC -> fp32 NCHW
// =====================================================================
__global__ __launch_bounds__(256)
void k_dec3(const h16* __restrict__ d2, const float* __restrict__ w,
            const float* __restrict__ b, float* __restrict__ out)
{
  long tid = (long)blockIdx.x * blockDim.x + threadIdx.x;
  if (tid >= (long)64 * 128 * 128 * 3) return;
  int co = (int)(tid % 3);
  long m = tid / 3;
  int ow = (int)(m & 127);
  int oh = (int)((m >> 7) & 127);
  int bb = (int)(m >> 14);
  float acc = b[co];
  #pragma unroll
  for (int kh = 0; kh < 3; ++kh) {
    int ih = oh - 1 + kh;
    if ((unsigned)ih >= 128u) continue;
    #pragma unroll
    for (int kw = 0; kw < 3; ++kw) {
      int iw = ow - 1 + kw;
      if ((unsigned)iw >= 128u) continue;
      const h16* src = d2 + (((long)bb * 128 + ih) * 128 + iw) * 64;
      const float* ww = w + (((long)co * 64) * 3 + kh) * 3 + kw;
      #pragma unroll 8
      for (int ci = 0; ci < 64; ++ci)
        acc += (float)src[ci] * ww[(long)ci * 9];
    }
  }
  out[(((long)bb * 3 + co) * 128 + oh) * 128 + ow] = tanhf(acc);
}

// =====================================================================
extern "C" void kernel_launch(void* const* d_in, const int* in_sizes, int n_in,
                              void* d_out, int out_size, void* d_ws, size_t ws_size,
                              hipStream_t stream)
{
  (void)in_sizes; (void)n_in; (void)out_size; (void)ws_size;
  const float* x      = (const float*)d_in[0];
  const float* enc_w1 = (const float*)d_in[1];
  const float* enc_b1 = (const float*)d_in[2];
  const float* enc_w2 = (const float*)d_in[3];
  const float* enc_b2 = (const float*)d_in[4];
  const float* enc_w3 = (const float*)d_in[5];
  const float* enc_b3 = (const float*)d_in[6];
  const float* emb    = (const float*)d_in[7];
  const float* dec_w1 = (const float*)d_in[8];
  const float* dec_b1 = (const float*)d_in[9];
  const float* dec_w2 = (const float*)d_in[10];
  const float* dec_b2 = (const float*)d_in[11];
  const float* dec_w3 = (const float*)d_in[12];
  const float* dec_b3 = (const float*)d_in[13];
  float* out = (float*)d_out;

  char* ws = (char*)d_ws;
  size_t off = 0;
  auto take = [&](size_t bytes) -> char* {
    char* p = ws + off;
    off = (off + bytes + 255) & ~(size_t)255;
    return p;
  };
  h16*   wE2    = (h16*)take((size_t)1024 * 128 * 2);
  h16*   wE3    = (h16*)take((size_t)1152 * 128 * 2);
  h16*   wD1    = (h16*)take((size_t)4 * 512 * 128 * 2);
  h16*   wD2    = (h16*)take((size_t)4 * 512 * 64 * 2);
  h16*   embF   = (h16*)take((size_t)128 * 512 * 2);
  float* enorm  = (float*)take((size_t)512 * 4);
  h16*   z_e    = (h16*)take((size_t)65536 * 128 * 2);
  h16*   h2zq   = (h16*)take((size_t)65536 * 128 * 2);          // h2, then z_q
  h16*   h1d1   = (h16*)take((size_t)64 * 64 * 64 * 128 * 2);   // h1, then d1
  char*  big    = take((size_t)65536 * 512 * 4);                // scores, then d2
  float* scores = (float*)big;
  h16*   d2buf  = (h16*)big;
  h16 *h1 = h1d1, *d1 = h1d1, *h2 = h2zq, *zq = h2zq;

  const int kl[2][2] = { {1, 3}, {0, 2} };   // output parity -> valid taps

  // ---- weight / codebook transforms (fragment-swizzled) ----
  k_wt_conv_frag<<<(1024 * 128 + 255) / 256, 256, 0, stream>>>(enc_w2, wE2, 128, 64, 4, 4);
  k_wt_conv_frag<<<(1152 * 128 + 255) / 256, 256, 0, stream>>>(enc_w3, wE3, 128, 128, 3, 3);
  for (int pa = 0; pa < 2; ++pa)
    for (int pb = 0; pb < 2; ++pb) {
      int ph = pa * 2 + pb;
      k_wt_deconv_frag<<<(512 * 128 + 255) / 256, 256, 0, stream>>>(
          dec_w1, wD1 + (size_t)ph * 512 * 128, 128, 128, 4, 4,
          kl[pa][0], kl[pa][1], kl[pb][0], kl[pb][1]);
      k_wt_deconv_frag<<<(512 * 64 + 255) / 256, 256, 0, stream>>>(
          dec_w2, wD2 + (size_t)ph * 512 * 64, 128, 64, 4, 4,
          kl[pa][0], kl[pa][1], kl[pb][0], kl[pb][1]);
    }
  k_embT2_frag<<<(128 * 512 + 255) / 256, 256, 0, stream>>>(emb, embF);
  k_enorm<<<2, 256, 0, stream>>>(emb, enorm);

  // ---- encoder ----
  k_enc1<<<65536, 256, 0, stream>>>(x, enc_w1, enc_b1, h1);
  // enc2: 64->128 k4 s2 p1, M=65536 K=1024 N=128  (grid: M/128, Ntiles/NT)
  k_wmma_conv<<<dim3(512, 2), 128, 0, stream>>>(h1, wE2, enc_b2, h2,
      64, 64, 6, 32, 32, 128, 4, 4, 2, 1, 8, 1);
  // enc3: 128->128 k3 s1 p1, M=65536 K=1152 N=128
  k_wmma_conv<<<dim3(512, 2), 128, 0, stream>>>(h2, wE3, enc_b3, z_e,
      32, 32, 7, 32, 32, 128, 3, 3, 1, 1, 8, 0);

  // ---- vector quantize ----
  k_vq_scores<<<dim3(512, 8), 128, 0, stream>>>(z_e, embF, enorm, scores);
  k_argmin_gather<<<8192, 256, 0, stream>>>(scores, emb, zq, 65536);

  // ---- decoder ----
  for (int pa = 0; pa < 2; ++pa)
    for (int pb = 0; pb < 2; ++pb) {
      int ph = pa * 2 + pb;
      // dec1: 128->128, out 64x64; per phase M=65536 K=512 N=128
      k_wmma_deconv<<<dim3(512, 2), 128, 0, stream>>>(zq,
          wD1 + (size_t)ph * 512 * 128, dec_b1, d1,
          32, 32, 7, 64, 64, 128,
          pa, pb, kl[pa][0], kl[pa][1], kl[pb][0], kl[pb][1], 8, 1);
    }
  for (int pa = 0; pa < 2; ++pa)
    for (int pb = 0; pb < 2; ++pb) {
      int ph = pa * 2 + pb;
      // dec2: 128->64, out 128x128; per phase M=262144 K=512 N=64
      k_wmma_deconv<<<dim3(2048, 1), 128, 0, stream>>>(d1,
          wD2 + (size_t)ph * 512 * 64, dec_b2, d2buf,
          64, 64, 7, 128, 128, 64,
          pa, pb, kl[pa][0], kl[pa][1], kl[pb][0], kl[pb][1], 4, 1);
    }
  k_dec3<<<12288, 256, 0, stream>>>(d2buf, dec_w3, dec_b3, out);
}
